// TransformerBlock_75531294868060
// MI455X (gfx1250) — matrix operand order repaired and checked by reading, not compiled, not run on hardware
//
#include <hip/hip_runtime.h>
#include <math.h>

#define BB   2
#define TT   2048
#define DD   1024
#define HH   16
#define DH   64
#define DFF  4096
#define MTOT (BB*TT)   // 4096 rows

typedef __bf16 bf16_t;
typedef bf16_t v16bf __attribute__((ext_vector_type(16)));
typedef float  v8f   __attribute__((ext_vector_type(8)));

union FragAB { v16bf h; unsigned u[8]; };
union FragC  { v8f v; float f[8]; };

__device__ __forceinline__ unsigned short f2bf(float f) {
  unsigned u = __builtin_bit_cast(unsigned, f);
  unsigned r = u + 0x7fffu + ((u >> 16) & 1u);   // round-to-nearest-even
  return (unsigned short)(r >> 16);
}

// ---------------------------------------------------------------- convert
__global__ __launch_bounds__(256) void cvt_f32_to_bf16(
    const float* __restrict__ in, unsigned short* __restrict__ out, int n) {
  int i = blockIdx.x * blockDim.x + threadIdx.x;
  int stride = gridDim.x * blockDim.x;
  for (; i < n; i += stride) out[i] = f2bf(in[i]);
}

// ---------------------------------------------------------------- layernorm
// matches reference: (x - mean) / (std + eps) * scale + bias, std = sqrt(pop var)
__global__ __launch_bounds__(256) void layernorm_bf16(
    const float* __restrict__ x, const float* __restrict__ sc,
    const float* __restrict__ bi, unsigned short* __restrict__ out) {
  __shared__ float s1[8], s2[8];
  const int row = blockIdx.x;
  const float* xr = x + (size_t)row * DD;
  float v0 = xr[threadIdx.x];
  float v1 = xr[threadIdx.x + 256];
  float v2 = xr[threadIdx.x + 512];
  float v3 = xr[threadIdx.x + 768];
  float sum = v0 + v1 + v2 + v3;
  float sq  = v0*v0 + v1*v1 + v2*v2 + v3*v3;
#pragma unroll
  for (int m = 16; m >= 1; m >>= 1) {
    sum += __shfl_xor(sum, m, 32);
    sq  += __shfl_xor(sq,  m, 32);
  }
  const int wave = threadIdx.x >> 5;
  if ((threadIdx.x & 31) == 0) { s1[wave] = sum; s2[wave] = sq; }
  __syncthreads();
  float ts = 0.f, tq = 0.f;
#pragma unroll
  for (int i = 0; i < 8; ++i) { ts += s1[i]; tq += s2[i]; }
  const float mean = ts * (1.f / DD);
  float var = tq * (1.f / DD) - mean * mean;
  var = var < 0.f ? 0.f : var;
  const float inv = 1.f / (sqrtf(var) + 1e-5f);
  unsigned short* orow = out + (size_t)row * DD;
  const int c = threadIdx.x;
  orow[c]       = f2bf((v0 - mean) * inv * sc[c]       + bi[c]);
  orow[c + 256] = f2bf((v1 - mean) * inv * sc[c + 256] + bi[c + 256]);
  orow[c + 512] = f2bf((v2 - mean) * inv * sc[c + 512] + bi[c + 512]);
  orow[c + 768] = f2bf((v3 - mean) * inv * sc[c + 768] + bi[c + 768]);
}

// ---------------------------------------------------------------- GEMM (bf16 WMMA)
// C[M,N] = A[M,K] * B[K,N] (+ epilogue). Block tile 128x64, 8 waves of 32x32.
// Double-buffered LDS: global loads for tile t+1 overlap the WMMAs of tile t.
// EPI: 0 plain->bf16, 1 +bias+residual->f32, 2 +bias+GELU->bf16, 3 *0.125->bf16 (Q)
template <int EPI>
__global__ __launch_bounds__(256) void gemm_bf16_wmma(
    const unsigned short* __restrict__ A, const unsigned short* __restrict__ Bw,
    const float* __restrict__ bias, const float* __restrict__ resid,
    void* __restrict__ outp, int M, int N, int K) {
  __shared__ unsigned short sA[2][128 * 40];   // 128 rows x 32 k, stride 40
  __shared__ unsigned short sB[2][64 * 40];    // transposed: 64 n x 32 k, stride 40
  (void)M;

  const int bm = blockIdx.y * 128;
  const int bn = blockIdx.x * 64;
  const int tid  = threadIdx.x;
  const int lane = tid & 31, wave = tid >> 5;
  const int half = lane >> 4, l15 = lane & 15;
  const int wm = wave & 3, wn = wave >> 2;
  constexpr int kkA[8] = {0, 2, 4, 6, 8, 10, 12, 14};  // contiguous A-frag K pairs (k = 16*half + s)

  // per-thread staging coordinates
  const int arow = tid >> 2;            // 0..63 (and +64)
  const int ag   = (tid & 3) * 8;       // 0,8,16,24
  const int bk_  = tid >> 3;            // 0..31
  const int bg   = (tid & 7) * 8;       // 0..56

  FragC acc[2][2];
#pragma unroll
  for (int i = 0; i < 2; ++i)
#pragma unroll
    for (int j = 0; j < 2; ++j)
#pragma unroll
      for (int r = 0; r < 8; ++r) acc[i][j].f[r] = 0.f;

  // prologue: stage tile 0 into buffer 0
  {
    uint4 a0 = *(const uint4*)(A + (size_t)(bm + arow) * K + ag);
    uint4 a1 = *(const uint4*)(A + (size_t)(bm + arow + 64) * K + ag);
    uint4 b0 = *(const uint4*)(Bw + (size_t)bk_ * N + bn + bg);
    *(uint4*)&sA[0][arow * 40 + ag] = a0;
    *(uint4*)&sA[0][(arow + 64) * 40 + ag] = a1;
    const unsigned short* pv = (const unsigned short*)&b0;
#pragma unroll
    for (int e = 0; e < 8; ++e) sB[0][(bg + e) * 40 + bk_] = pv[e];
  }
  __syncthreads();

  const int nk = K >> 5;
  for (int kt = 0; kt < nk; ++kt) {
    const int cur = kt & 1;
    const bool has_next = (kt + 1 < nk);

    // issue next-tile global loads early (latency overlapped by WMMAs below)
    uint4 na0, na1, nb0;
    if (has_next) {
      const int k0n = (kt + 1) << 5;
      na0 = *(const uint4*)(A + (size_t)(bm + arow) * K + k0n + ag);
      na1 = *(const uint4*)(A + (size_t)(bm + arow + 64) * K + k0n + ag);
      nb0 = *(const uint4*)(Bw + (size_t)(k0n + bk_) * N + bn + bg);
      if (kt + 2 < nk) {  // L2 prefetch two tiles ahead
        __builtin_prefetch(A + (size_t)(bm + arow) * K + ((kt + 2) << 5) + ag, 0, 0);
        __builtin_prefetch(Bw + (size_t)(((kt + 2) << 5) + bk_) * N + bn, 0, 0);
      }
    }

    // fragments from current buffer, then 4 WMMAs (waits coalesce)
    FragAB a[2], b[2];
#pragma unroll
    for (int mi = 0; mi < 2; ++mi) {
      int row = (wm * 32 + mi * 16 + l15) * 40 + 16 * half;
#pragma unroll
      for (int j = 0; j < 8; ++j) a[mi].u[j] = *(const unsigned*)&sA[cur][row + kkA[j]];
    }
#pragma unroll
    for (int ni = 0; ni < 2; ++ni) {
      int row = (wn * 32 + ni * 16 + l15) * 40 + 16 * half;
#pragma unroll
      for (int j = 0; j < 8; ++j) b[ni].u[j] = *(const unsigned*)&sB[cur][row + 2 * j];
    }
#pragma unroll
    for (int mi = 0; mi < 2; ++mi)
#pragma unroll
      for (int ni = 0; ni < 2; ++ni)
        acc[mi][ni].v = __builtin_amdgcn_wmma_f32_16x16x32_bf16(
            false, a[mi].h, false, b[ni].h, (short)0, acc[mi][ni].v, false, false);

    // commit next tile to the alternate buffer
    if (has_next) {
      *(uint4*)&sA[1 - cur][arow * 40 + ag] = na0;
      *(uint4*)&sA[1 - cur][(arow + 64) * 40 + ag] = na1;
      const unsigned short* pv = (const unsigned short*)&nb0;
#pragma unroll
      for (int e = 0; e < 8; ++e) sB[1 - cur][(bg + e) * 40 + bk_] = pv[e];
      __syncthreads();
    }
  }

  // epilogue
#pragma unroll
  for (int mi = 0; mi < 2; ++mi)
#pragma unroll
    for (int ni = 0; ni < 2; ++ni)
#pragma unroll
      for (int r = 0; r < 8; ++r) {
        int row = bm + wm * 32 + mi * 16 + r + 8 * half;
        int col = bn + wn * 32 + ni * 16 + l15;
        float v = acc[mi][ni].f[r];
        if (EPI == 1 || EPI == 2) v += bias[col];
        if (EPI == 2) v = 0.5f * v * (1.f + erff(v * 0.70710678118654752f));
        if (EPI == 3) v *= 0.125f;  // fold attention 1/sqrt(Dh) into Q (exact in bf16)
        if (EPI == 1) {
          v += resid[(size_t)row * N + col];
          ((float*)outp)[(size_t)row * N + col] = v;
        } else {
          ((unsigned short*)outp)[(size_t)row * N + col] = f2bf(v);
        }
      }
}

// ---------------------------------------------------------------- flash attention
// One block = one (b,h) x 64-query tile; 4 waves, each owns 16 query rows.
// Q is pre-scaled by 1/sqrt(Dh). All B-fragments preloaded before each WMMA burst.
__global__ __launch_bounds__(128) void attn_flash_wmma(
    const unsigned short* __restrict__ qb, const unsigned short* __restrict__ kb,
    const unsigned short* __restrict__ vb, unsigned short* __restrict__ ob) {
  __shared__ unsigned short sK[64 * 72];  // K tile row-major [key][d]  (= B^T for scores)
  __shared__ unsigned short sV[64 * 72];  // V tile transposed [d][key] (= B^T for PV)
  __shared__ unsigned short sP[64 * 72];  // per-wave P (bf16) staging

  const int b  = blockIdx.x >> 4;
  const int h  = blockIdx.x & 15;
  const int q0 = blockIdx.y * 64;
  const int tid  = threadIdx.x;
  const int lane = tid & 31, wave = tid >> 5;
  const int half = lane >> 4, l15 = lane & 15;
  constexpr int kkA[8] = {0, 2, 4, 6, 8, 10, 12, 14};
  const float NEG_INF = -__builtin_inff();

  // Q fragments: 16 rows/wave, K-dim 64 -> 2 frags, straight from global
  FragAB aq[2];
  {
    size_t qrow = (size_t)(b * TT + q0 + wave * 16 + l15) * DD + h * DH;
#pragma unroll
    for (int fi = 0; fi < 2; ++fi)
#pragma unroll
      for (int j = 0; j < 8; ++j)
        aq[fi].u[j] = *(const unsigned*)(qb + qrow + kkA[j] + 16 * half + 32 * fi);
  }

  float mrun[8], lrun[8];
  FragC o[4];
#pragma unroll
  for (int r = 0; r < 8; ++r) { mrun[r] = NEG_INF; lrun[r] = 0.f; }
#pragma unroll
  for (int nt = 0; nt < 4; ++nt)
#pragma unroll
    for (int r = 0; r < 8; ++r) o[nt].f[r] = 0.f;

  const int nkt = q0 / 64 + 1;  // causal: only key tiles up to the diagonal
  for (int kt = 0; kt < nkt; ++kt) {
    // stage K (row-major) and V (transposed)
#pragma unroll
    for (int i = 0; i < 4; ++i) {
      int idx = tid + 128 * i;
      int row = idx >> 3;
      int g   = (idx & 7) * 8;
      size_t src = (size_t)(b * TT + kt * 64 + row) * DD + h * DH + g;
      uint4 kv = *(const uint4*)(kb + src);
      const uint2* k2 = (const uint2*)&kv;
      *(uint2*)&sK[row * 72 + g]     = k2[0];
      *(uint2*)&sK[row * 72 + g + 4] = k2[1];
      uint4 vv = *(const uint4*)(vb + src);
      const unsigned short* pv = (const unsigned short*)&vv;
#pragma unroll
      for (int e = 0; e < 8; ++e) sV[(g + e) * 72 + row] = pv[e];
    }
    __syncthreads();

    // preload all 8 K^T fragments, then 8 WMMAs back-to-back
    FragAB bk[8];
#pragma unroll
    for (int nt = 0; nt < 4; ++nt)
#pragma unroll
      for (int fi = 0; fi < 2; ++fi) {
        int row = (nt * 16 + l15) * 72 + 16 * half + 32 * fi;
#pragma unroll
        for (int j = 0; j < 8; ++j)
          bk[nt * 2 + fi].u[j] = *(const unsigned*)&sK[row + 2 * j];
      }
    FragC s[4];
#pragma unroll
    for (int nt = 0; nt < 4; ++nt) {
#pragma unroll
      for (int r = 0; r < 8; ++r) s[nt].f[r] = 0.f;
      s[nt].v = __builtin_amdgcn_wmma_f32_16x16x32_bf16(
          false, aq[0].h, false, bk[nt * 2].h, (short)0, s[nt].v, false, false);
      s[nt].v = __builtin_amdgcn_wmma_f32_16x16x32_bf16(
          false, aq[1].h, false, bk[nt * 2 + 1].h, (short)0, s[nt].v, false, false);
    }

    // causal mask: only the diagonal tile can mask anything
    if (kt == nkt - 1) {
#pragma unroll
      for (int nt = 0; nt < 4; ++nt)
#pragma unroll
        for (int r = 0; r < 8; ++r) {
          int kg = kt * 64 + nt * 16 + l15;
          int qg = q0 + wave * 16 + r + 8 * half;
          if (kg > qg) s[nt].f[r] = NEG_INF;
        }
    }

    // online softmax update (row stats per C-fragment VGPR, reduced over 16 lanes)
    float alpha[8], mnew[8];
#pragma unroll
    for (int r = 0; r < 8; ++r) {
      float mx = fmaxf(fmaxf(s[0].f[r], s[1].f[r]), fmaxf(s[2].f[r], s[3].f[r]));
#pragma unroll
      for (int m = 8; m >= 1; m >>= 1) mx = fmaxf(mx, __shfl_xor(mx, m, 32));
      mnew[r]  = fmaxf(mrun[r], mx);
      alpha[r] = (mrun[r] == NEG_INF) ? 0.f : __expf(mrun[r] - mnew[r]);
    }
#pragma unroll
    for (int nt = 0; nt < 4; ++nt)
#pragma unroll
      for (int r = 0; r < 8; ++r) s[nt].f[r] = __expf(s[nt].f[r] - mnew[r]);
#pragma unroll
    for (int r = 0; r < 8; ++r) {
      float ls = s[0].f[r] + s[1].f[r] + s[2].f[r] + s[3].f[r];
#pragma unroll
      for (int m = 8; m >= 1; m >>= 1) ls += __shfl_xor(ls, m, 32);
      lrun[r] = lrun[r] * alpha[r] + ls;
      mrun[r] = mnew[r];
    }
#pragma unroll
    for (int nt = 0; nt < 4; ++nt)
#pragma unroll
      for (int r = 0; r < 8; ++r) o[nt].f[r] *= alpha[r];

    // P (C-layout f32) -> LDS bf16 -> A-fragment layout (per-wave private region)
#pragma unroll
    for (int nt = 0; nt < 4; ++nt)
#pragma unroll
      for (int r = 0; r < 8; ++r)
        sP[(wave * 16 + r + 8 * half) * 72 + nt * 16 + l15] = f2bf(s[nt].f[r]);

    FragAB ap[2];
    {
      int row = (wave * 16 + l15) * 72 + 16 * half;
#pragma unroll
      for (int fi = 0; fi < 2; ++fi)
#pragma unroll
        for (int j = 0; j < 8; ++j)
          ap[fi].u[j] = *(const unsigned*)&sP[row + kkA[j] + 32 * fi];
    }

    // preload all 8 V fragments, then 8 WMMAs back-to-back
    FragAB bv[8];
#pragma unroll
    for (int nt = 0; nt < 4; ++nt)
#pragma unroll
      for (int fi = 0; fi < 2; ++fi) {
        int row = (nt * 16 + l15) * 72 + 16 * half + 32 * fi;
#pragma unroll
        for (int j = 0; j < 8; ++j)
          bv[nt * 2 + fi].u[j] = *(const unsigned*)&sV[row + 2 * j];
      }
#pragma unroll
    for (int nt = 0; nt < 4; ++nt) {
      o[nt].v = __builtin_amdgcn_wmma_f32_16x16x32_bf16(
          false, ap[0].h, false, bv[nt * 2].h, (short)0, o[nt].v, false, false);
      o[nt].v = __builtin_amdgcn_wmma_f32_16x16x32_bf16(
          false, ap[1].h, false, bv[nt * 2 + 1].h, (short)0, o[nt].v, false, false);
    }
    __syncthreads();
  }

  // normalize and store bf16 (row-major [B*T, D], head-column block)
#pragma unroll
  for (int nt = 0; nt < 4; ++nt)
#pragma unroll
    for (int r = 0; r < 8; ++r) {
      size_t row = (size_t)(b * TT + q0 + wave * 16 + r + 8 * half);
      int col = h * DH + nt * 16 + l15;
      ob[row * DD + col] = f2bf(o[nt].f[r] / lrun[r]);
    }
}

// ---------------------------------------------------------------- host
extern "C" void kernel_launch(void* const* d_in, const int* in_sizes, int n_in,
                              void* d_out, int out_size, void* d_ws, size_t ws_size,
                              hipStream_t stream) {
  (void)in_sizes; (void)n_in; (void)out_size; (void)ws_size;
  const float* x    = (const float*)d_in[0];
  const float* wq   = (const float*)d_in[1];
  const float* wk   = (const float*)d_in[2];
  const float* wv   = (const float*)d_in[3];
  const float* wo   = (const float*)d_in[4];
  const float* bo   = (const float*)d_in[5];
  const float* w1   = (const float*)d_in[6];
  const float* b1   = (const float*)d_in[7];
  const float* w2   = (const float*)d_in[8];
  const float* b2   = (const float*)d_in[9];
  const float* ln1s = (const float*)d_in[10];
  const float* ln1b = (const float*)d_in[11];
  const float* ln2s = (const float*)d_in[12];
  const float* ln2b = (const float*)d_in[13];

  char* ws = (char*)d_ws;
  size_t off = 0;
  auto take = [&](size_t bytes) {
    char* p = ws + off;
    off = (off + bytes + 255) & ~(size_t)255;
    return p;
  };
  unsigned short* wqb = (unsigned short*)take((size_t)DD * DD * 2);
  unsigned short* wkb = (unsigned short*)take((size_t)DD * DD * 2);
  unsigned short* wvb = (unsigned short*)take((size_t)DD * DD * 2);
  unsigned short* wob = (unsigned short*)take((size_t)DD * DD * 2);
  unsigned short* w1b = (unsigned short*)take((size_t)DD * DFF * 2);
  unsigned short* w2b = (unsigned short*)take((size_t)DFF * DD * 2);
  unsigned short* hb  = (unsigned short*)take((size_t)MTOT * DD * 2);   // LN out (reused for LN2)
  unsigned short* qbf = (unsigned short*)take((size_t)MTOT * DD * 2);
  unsigned short* kbf = (unsigned short*)take((size_t)MTOT * DD * 2);
  unsigned short* vbf = (unsigned short*)take((size_t)MTOT * DD * 2);
  unsigned short* aob = (unsigned short*)take((size_t)MTOT * DD * 2);   // attention out
  float*          x1  = (float*)take((size_t)MTOT * DD * 4);            // residual after attn
  unsigned short* ffb = (unsigned short*)take((size_t)MTOT * DFF * 2);  // GELU(h2 W1 + b1)

  // weights fp32 -> bf16 (deterministic, every call)
  cvt_f32_to_bf16<<<512, 256, 0, stream>>>(wq, wqb, DD * DD);
  cvt_f32_to_bf16<<<512, 256, 0, stream>>>(wk, wkb, DD * DD);
  cvt_f32_to_bf16<<<512, 256, 0, stream>>>(wv, wvb, DD * DD);
  cvt_f32_to_bf16<<<512, 256, 0, stream>>>(wo, wob, DD * DD);
  cvt_f32_to_bf16<<<1024, 256, 0, stream>>>(w1, w1b, DD * DFF);
  cvt_f32_to_bf16<<<1024, 256, 0, stream>>>(w2, w2b, DFF * DD);

  // LN1
  layernorm_bf16<<<MTOT, 256, 0, stream>>>(x, ln1s, ln1b, hb);

  // Q (pre-scaled by 1/sqrt(Dh)), K, V projections
  dim3 gD(DD / 64, MTOT / 128);
  gemm_bf16_wmma<3><<<gD, 256, 0, stream>>>(hb, wqb, nullptr, nullptr, qbf, MTOT, DD, DD);
  gemm_bf16_wmma<0><<<gD, 256, 0, stream>>>(hb, wkb, nullptr, nullptr, kbf, MTOT, DD, DD);
  gemm_bf16_wmma<0><<<gD, 256, 0, stream>>>(hb, wvb, nullptr, nullptr, vbf, MTOT, DD, DD);

  // causal flash attention
  attn_flash_wmma<<<dim3(BB * HH, TT / 64), 128, 0, stream>>>(qbf, kbf, vbf, aob);

  // x1 = x + attn_out @ wo + bo
  gemm_bf16_wmma<1><<<gD, 256, 0, stream>>>(aob, wob, bo, x, x1, MTOT, DD, DD);

  // LN2
  layernorm_bf16<<<MTOT, 256, 0, stream>>>(x1, ln2s, ln2b, hb);

  // ff = GELU(h2 @ w1 + b1)
  gemm_bf16_wmma<2><<<dim3(DFF / 64, MTOT / 128), 256, 0, stream>>>(
      hb, w1b, b1, nullptr, ffb, MTOT, DFF, DD);

  // out = x1 + ff @ w2 + b2
  gemm_bf16_wmma<1><<<gD, 256, 0, stream>>>(ffb, w2b, b2, x1, d_out, MTOT, DD, DFF);
}
